// StateTransformerBlock_20229295964204
// MI455X (gfx1250) — matrix operand order repaired and checked by reading, not compiled, not run on hardware
//
#include <hip/hip_runtime.h>

typedef _Float16 h16;
typedef _Float16 v16h __attribute__((ext_vector_type(16)));
typedef _Float16 v8h  __attribute__((ext_vector_type(8)));
typedef float    v8f  __attribute__((ext_vector_type(8)));

#define DV 1024
#define HV 8
#define FV 4096
#define TV 2048
#define SV 512
#define BV 4

// ---------------------------------------------------------------------------
// Generic batched WMMA GEMM:  C[z] = A[z] (MxK) * B[z]^T (NxK rows) + bias
// Both A and B are f16 row-major with the reduction (K) axis contiguous.
// Block: 256 threads = 8 waves, block tile 128(M) x 64(N), wave tile 32x32.
// ---------------------------------------------------------------------------
struct GemmP {
    const h16* A; const h16* B; void* C; const float* bias;
    int K, lda, ldb, ldc;
    long long sAb, sAh, sBb, sBh, sCb, sCh;   // batch strides (elements)
    int zInner; long long biasH;              // z = zb*zInner + zh
    float alpha; int transT, transD;          // mode-3 scale / mode-4 transpose
};

__device__ __forceinline__ v16h ldA(const h16* p) {
    // A fragment, 16x32 f16: elems[0..15] = K+16*half .. +15
    return *(const v16h*)p;
}

template <int MODE>
__device__ __forceinline__ void storeTile(const GemmP& p, int zb, int zh, v8f acc,
                                          int mBase, int nc, int half, const float* bias) {
    const float bv = (MODE == 0 || MODE == 1 || MODE == 2 || MODE == 4) ? bias[nc] : 0.f;
#pragma unroll
    for (int r = 0; r < 8; ++r) {
        const int mr = mBase + r + 8 * half;     // C layout: VGPR r, lanes16-31 -> M+8
        float v = acc[r];
        if (MODE == 0) {                         // f32 + bias
            float* C = (float*)p.C + zb * p.sCb + zh * p.sCh;
            C[(long long)mr * p.ldc + nc] = v + bv;
        } else if (MODE == 1) {                  // f16 + bias
            h16* C = (h16*)p.C + zb * p.sCb + zh * p.sCh;
            C[(long long)mr * p.ldc + nc] = (h16)(v + bv);
        } else if (MODE == 2) {                  // f16 + bias + relu
            h16* C = (h16*)p.C + zb * p.sCb + zh * p.sCh;
            float t = v + bv;
            C[(long long)mr * p.ldc + nc] = (h16)(t > 0.f ? t : 0.f);
        } else if (MODE == 3) {                  // f16 * alpha (scores / ctx)
            h16* C = (h16*)p.C + zb * p.sCb + zh * p.sCh;
            C[(long long)mr * p.ldc + nc] = (h16)(v * p.alpha);
        } else {                                 // MODE 4: f16 + bias, transposed (v -> vT[b][e][t])
            h16* C = (h16*)p.C + zb * p.sCb + zh * p.sCh;
            long long b = mr / p.transT, t = mr % p.transT;
            long long idx = b * (long long)p.transD * p.transT + (long long)nc * p.transT + t;
            C[idx] = (h16)(v + bv);
        }
    }
}

template <int MODE>
__global__ __launch_bounds__(256) void gemm_wmma(GemmP p) {
    const int lane = threadIdx.x & 31;
    const int wave = threadIdx.x >> 5;
    const int wm = wave >> 1, wn = wave & 1;
    const int z  = blockIdx.z;
    const int zb = z / p.zInner, zh = z % p.zInner;

    const h16* A  = p.A + zb * p.sAb + zh * p.sAh;
    const h16* Bm = p.B + zb * p.sBb + zh * p.sBh;

    const int m0   = blockIdx.y * 128 + wm * 32;
    const int n0   = blockIdx.x * 64  + wn * 32;
    const int row  = lane & 15;
    const int half = lane >> 4;

    const h16* a0 = A  + (long long)(m0 + row) * p.lda + 16 * half;
    const h16* a1 = a0 + 16LL * p.lda;
    const h16* b0 = Bm + (long long)(n0 + row) * p.ldb + 16 * half;
    const h16* b1 = b0 + 16LL * p.ldb;

    v8f c00 = {0.f,0.f,0.f,0.f,0.f,0.f,0.f,0.f};
    v8f c01 = c00, c10 = c00, c11 = c00;

    for (int k = 0; k < p.K; k += 32) {
        __builtin_prefetch(a0 + k + 256, 0, 0);   // global_prefetch_b8 lookahead
        __builtin_prefetch(b0 + k + 256, 0, 0);
        v16h fa0 = ldA(a0 + k);
        v16h fa1 = ldA(a1 + k);
        v16h fb0 = *(const v16h*)(b0 + k);
        v16h fb1 = *(const v16h*)(b1 + k);
        c00 = __builtin_amdgcn_wmma_f32_16x16x32_f16(false, fa0, false, fb0, (short)0, c00, false, false);
        c01 = __builtin_amdgcn_wmma_f32_16x16x32_f16(false, fa0, false, fb1, (short)0, c01, false, false);
        c10 = __builtin_amdgcn_wmma_f32_16x16x32_f16(false, fa1, false, fb0, (short)0, c10, false, false);
        c11 = __builtin_amdgcn_wmma_f32_16x16x32_f16(false, fa1, false, fb1, (short)0, c11, false, false);
    }

    const float* bb = p.bias ? (p.bias + zh * p.biasH) : nullptr;
    storeTile<MODE>(p, zb, zh, c00, m0,      n0 + row,      half, bb);
    storeTile<MODE>(p, zb, zh, c01, m0,      n0 + 16 + row, half, bb);
    storeTile<MODE>(p, zb, zh, c10, m0 + 16, n0 + row,      half, bb);
    storeTile<MODE>(p, zb, zh, c11, m0 + 16, n0 + 16 + row, half, bb);
}

// ---------------------------------------------------------------------------
// Elementwise / reduction kernels
// ---------------------------------------------------------------------------
__global__ __launch_bounds__(256) void build_store(const float* state, const float* input, h16* store) {
    long long i = (long long)blockIdx.x * 256 + threadIdx.x;   // B*T*D elems
    long long b = i / ((long long)TV * DV);
    long long r = i % ((long long)TV * DV);
    long long t = r / DV, d = r % DV;
    float v = (t < SV) ? state[(b * SV + t) * DV + d]
                       : input[(b * (TV - SV) + (t - SV)) * DV + d];
    store[i] = (h16)v;
}

// in [G,K,N] f32 -> out [G,N,K] f16 (tiled via LDS)
__global__ __launch_bounds__(256) void transpose_f16(const float* in, h16* out, int K, int N) {
    __shared__ float t[32][33];
    const long long g = blockIdx.z;
    const float* ig = in  + g * (long long)K * N;
    h16*         og = out + g * (long long)K * N;
    const int k0 = blockIdx.y * 32, n0 = blockIdx.x * 32;
    const int tx = threadIdx.x, ty = threadIdx.y;     // (32,8)
    for (int i = ty; i < 32; i += 8) t[i][tx] = ig[(long long)(k0 + i) * N + (n0 + tx)];
    __syncthreads();
    for (int i = ty; i < 32; i += 8) og[(long long)(n0 + i) * K + (k0 + tx)] = (h16)t[tx][i];
}

__global__ __launch_bounds__(256) void softmax_inplace(h16* s) {
    __shared__ float red[256];
    const int tid = threadIdx.x;
    long long base = (long long)blockIdx.x * TV;
    float v[8]; float mx = -1e30f;
#pragma unroll
    for (int i = 0; i < 8; ++i) { v[i] = (float)s[base + i * 256 + tid]; mx = fmaxf(mx, v[i]); }
    red[tid] = mx; __syncthreads();
    for (int o = 128; o > 0; o >>= 1) { if (tid < o) red[tid] = fmaxf(red[tid], red[tid + o]); __syncthreads(); }
    mx = red[0]; __syncthreads();
    float sum = 0.f;
#pragma unroll
    for (int i = 0; i < 8; ++i) { v[i] = __expf(v[i] - mx); sum += v[i]; }
    red[tid] = sum; __syncthreads();
    for (int o = 128; o > 0; o >>= 1) { if (tid < o) red[tid] += red[tid + o]; __syncthreads(); }
    const float inv = 1.f / red[0];
#pragma unroll
    for (int i = 0; i < 8; ++i) s[base + i * 256 + tid] = (h16)(v[i] * inv);
}

__global__ __launch_bounds__(256) void layernorm_k(const float* y, const float* resid,
                                                   const float* g, const float* b,
                                                   float* out32, h16* out16) {
    __shared__ float red[256];
    const int tid = threadIdx.x;
    long long base = (long long)blockIdx.x * DV;
    float v[4]; float s = 0.f;
#pragma unroll
    for (int i = 0; i < 4; ++i) { v[i] = y[base + i * 256 + tid] + resid[base + i * 256 + tid]; s += v[i]; }
    red[tid] = s; __syncthreads();
    for (int o = 128; o > 0; o >>= 1) { if (tid < o) red[tid] += red[tid + o]; __syncthreads(); }
    const float mean = red[0] * (1.f / DV); __syncthreads();
    float ss = 0.f;
#pragma unroll
    for (int i = 0; i < 4; ++i) { float d = v[i] - mean; ss += d * d; }
    red[tid] = ss; __syncthreads();
    for (int o = 128; o > 0; o >>= 1) { if (tid < o) red[tid] += red[tid + o]; __syncthreads(); }
    const float r = rsqrtf(red[0] * (1.f / DV) + 1e-6f);
#pragma unroll
    for (int i = 0; i < 4; ++i) {
        float xv = (v[i] - mean) * r * g[i * 256 + tid] + b[i * 256 + tid];
        out32[base + i * 256 + tid] = xv;
        if (out16) out16[base + i * 256 + tid] = (h16)xv;
    }
}

__global__ __launch_bounds__(256) void gated_mul(h16* t1, const h16* t2) {
    long long i = (long long)blockIdx.x * 256 + threadIdx.x;
    t1[i] = (h16)((float)t1[i] * (float)t2[i]);
}

// ---------------------------------------------------------------------------
// Host launcher
// ---------------------------------------------------------------------------
static GemmP mkP(const h16* A, const h16* B, void* C, const float* bias,
                 int K, int lda, int ldb, int ldc,
                 long long sAb, long long sAh, long long sBb, long long sBh,
                 long long sCb, long long sCh, int zInner, long long biasH,
                 float alpha, int transT, int transD) {
    GemmP p; p.A = A; p.B = B; p.C = C; p.bias = bias;
    p.K = K; p.lda = lda; p.ldb = ldb; p.ldc = ldc;
    p.sAb = sAb; p.sAh = sAh; p.sBb = sBb; p.sBh = sBh; p.sCb = sCb; p.sCh = sCh;
    p.zInner = zInner; p.biasH = biasH; p.alpha = alpha; p.transT = transT; p.transD = transD;
    return p;
}

extern "C" void kernel_launch(void* const* d_in, const int* in_sizes, int n_in,
                              void* d_out, int out_size, void* d_ws, size_t ws_size,
                              hipStream_t stream) {
    const float* state = (const float*)d_in[0];
    const float* input = (const float*)d_in[1];
    const float* Wk = (const float*)d_in[2];  const float* bk = (const float*)d_in[3];
    const float* Wv = (const float*)d_in[4];  const float* bv = (const float*)d_in[5];
    const float* Wq = (const float*)d_in[6];  const float* bq = (const float*)d_in[7];
    const float* Wo = (const float*)d_in[8];  const float* bo = (const float*)d_in[9];
    const float* ln1g = (const float*)d_in[10]; const float* ln1b = (const float*)d_in[11];
    const float* Wi = (const float*)d_in[12]; const float* bi = (const float*)d_in[13];
    const float* Wg = (const float*)d_in[14]; const float* bg = (const float*)d_in[15];
    const float* Wl = (const float*)d_in[16]; const float* bl = (const float*)d_in[17];
    const float* Wf = (const float*)d_in[18]; const float* bf = (const float*)d_in[19];
    const float* ln2g = (const float*)d_in[20]; const float* ln2b = (const float*)d_in[21];

    char* w = (char*)d_ws;
    auto alloc = [&](size_t bytes) -> char* { char* r = w; w += (bytes + 255) & ~(size_t)255; return r; };

    h16* WkT  = (h16*)alloc((size_t)DV * DV * 2);
    h16* WvT  = (h16*)alloc((size_t)DV * DV * 2);
    h16* WqT  = (h16*)alloc((size_t)HV * DV * DV * 2);
    h16* WoT  = (h16*)alloc((size_t)DV * HV * DV * 2);
    h16* WiT  = (h16*)alloc((size_t)FV * DV * 2);
    h16* WgT  = (h16*)alloc((size_t)FV * FV * 2);
    h16* WlT  = (h16*)alloc((size_t)FV * FV * 2);
    h16* WfT  = (h16*)alloc((size_t)DV * FV * 2);
    h16* sto  = (h16*)alloc((size_t)BV * TV * DV * 2);
    h16* kbuf = (h16*)alloc((size_t)BV * TV * DV * 2);
    h16* vT   = (h16*)alloc((size_t)BV * DV * TV * 2);
    h16* qbuf = (h16*)alloc((size_t)BV * HV * SV * DV * 2);
    h16* scr  = (h16*)alloc((size_t)BV * HV * SV * TV * 2);
    h16* cat  = (h16*)alloc((size_t)BV * SV * HV * DV * 2);
    float* ybuf = (float*)alloc((size_t)BV * SV * DV * 4);
    float* x32  = (float*)alloc((size_t)BV * SV * DV * 4);
    h16*   x16  = (h16*)alloc((size_t)BV * SV * DV * 2);
    h16* inner = (h16*)alloc((size_t)BV * SV * FV * 2);
    h16* t1    = (h16*)alloc((size_t)BV * SV * FV * 2);
    h16* t2    = (h16*)alloc((size_t)BV * SV * FV * 2);
    float* ffb = (float*)alloc((size_t)BV * SV * DV * 4);

    dim3 tb(32, 8);
    // store = f16 concat(state, input)
    build_store<<<(BV * TV * DV) / 256, 256, 0, stream>>>(state, input, sto);
    // weight transposes -> f16
    transpose_f16<<<dim3(32, 32, 1),  tb, 0, stream>>>(Wk, WkT, DV, DV);
    transpose_f16<<<dim3(32, 32, 1),  tb, 0, stream>>>(Wv, WvT, DV, DV);
    transpose_f16<<<dim3(32, 32, HV), tb, 0, stream>>>(Wq, WqT, DV, DV);
    transpose_f16<<<dim3(32, 256, 1), tb, 0, stream>>>(Wo, WoT, HV * DV, DV);
    transpose_f16<<<dim3(128, 32, 1), tb, 0, stream>>>(Wi, WiT, DV, FV);
    transpose_f16<<<dim3(128, 128, 1), tb, 0, stream>>>(Wg, WgT, FV, FV);
    transpose_f16<<<dim3(128, 128, 1), tb, 0, stream>>>(Wl, WlT, FV, FV);
    transpose_f16<<<dim3(32, 128, 1), tb, 0, stream>>>(Wf, WfT, FV, DV);

    // k = store @ Wk + bk   (f16 out)
    gemm_wmma<1><<<dim3(16, 64, 1), 256, 0, stream>>>(
        mkP(sto, WkT, kbuf, bk, DV, DV, DV, DV, 0,0,0,0,0,0, 1, 0, 1.f, 0, 0));
    // vT[b][e][t] = (store @ Wv + bv)^T
    gemm_wmma<4><<<dim3(16, 64, 1), 256, 0, stream>>>(
        mkP(sto, WvT, vT, bv, DV, DV, DV, 0, 0,0,0,0,0,0, 1, 0, 1.f, TV, DV));
    // q[b,h] = state @ Wq[h] + bq[h]   (A rows are the state prefix of store)
    gemm_wmma<1><<<dim3(16, 4, BV * HV), 256, 0, stream>>>(
        mkP(sto, WqT, qbuf, bq, DV, DV, DV, DV,
            (long long)TV * DV, 0, 0, (long long)DV * DV,
            (long long)HV * SV * DV, (long long)SV * DV, HV, DV, 1.f, 0, 0));
    // scores[b,h] = q[b,h] @ k[b]^T / 32   (f16, in-place softmax next)
    gemm_wmma<3><<<dim3(32, 4, BV * HV), 256, 0, stream>>>(
        mkP(qbuf, kbuf, scr, nullptr, DV, DV, DV, TV,
            (long long)HV * SV * DV, (long long)SV * DV, (long long)TV * DV, 0,
            (long long)HV * SV * TV, (long long)SV * TV, HV, 0, 0.03125f, 0, 0));
    softmax_inplace<<<BV * HV * SV, 256, 0, stream>>>(scr);
    // ctx[b,h] = attn[b,h] @ v[b]  -> written into concat layout [b,s,h*D+e]
    gemm_wmma<3><<<dim3(16, 4, BV * HV), 256, 0, stream>>>(
        mkP(scr, vT, cat, nullptr, TV, TV, TV, HV * DV,
            (long long)HV * SV * TV, (long long)SV * TV, (long long)DV * TV, 0,
            (long long)SV * HV * DV, (long long)DV, HV, 0, 1.f, 0, 0));
    // attn_out = concat @ Wo + bo (f32)
    gemm_wmma<0><<<dim3(16, 16, 1), 256, 0, stream>>>(
        mkP(cat, WoT, ybuf, bo, HV * DV, HV * DV, HV * DV, DV, 0,0,0,0,0,0, 1, 0, 1.f, 0, 0));
    // x = LN1(attn_out + state)
    layernorm_k<<<BV * SV, 256, 0, stream>>>(ybuf, state, ln1g, ln1b, x32, x16);
    // inner = relu(x @ Wi + bi)
    gemm_wmma<2><<<dim3(64, 16, 1), 256, 0, stream>>>(
        mkP(x16, WiT, inner, bi, DV, DV, DV, FV, 0,0,0,0,0,0, 1, 0, 1.f, 0, 0));
    // t1 = relu(inner @ Wg + bg); t2 = inner @ Wl + bl
    gemm_wmma<2><<<dim3(64, 16, 1), 256, 0, stream>>>(
        mkP(inner, WgT, t1, bg, FV, FV, FV, FV, 0,0,0,0,0,0, 1, 0, 1.f, 0, 0));
    gemm_wmma<1><<<dim3(64, 16, 1), 256, 0, stream>>>(
        mkP(inner, WlT, t2, bl, FV, FV, FV, FV, 0,0,0,0,0,0, 1, 0, 1.f, 0, 0));
    gated_mul<<<(BV * SV * FV) / 256, 256, 0, stream>>>(t1, t2);
    // ff = gated @ Wf + bf (f32)
    gemm_wmma<0><<<dim3(16, 16, 1), 256, 0, stream>>>(
        mkP(t1, WfT, ffb, bf, FV, FV, FV, DV, 0,0,0,0,0,0, 1, 0, 1.f, 0, 0));
    // out = LN2(ff + x)
    layernorm_k<<<BV * SV, 256, 0, stream>>>(ffb, x32, ln2g, ln2b, (float*)d_out, nullptr);
}
